// BiLSTMDecoderModel_24807731101752
// MI455X (gfx1250) — matrix operand order repaired and checked by reading, not compiled, not run on hardware
//
#include <hip/hip_runtime.h>
#include <hip/hip_bf16.h>

// ---------------------------------------------------------------------------
// BiLSTM decoder for MI455X (gfx1250, wave32, WMMA), fused step kernel.
//   emb = tanh(embed[seq])                      [S,B,E] f16
//   forward LSTM: 256 sequential fused steps:
//       per WG: 2(M)x4(gate) waves of 32x32 WMMA tiles (f16 -> f32 acc),
//       gate exchange via LDS, in-kernel sigmoid/tanh cell update.
//   backward LSTM: exactly ONE step on emb[0] (only b_hs[0] is consumed)
//   logits = [h_f, h_b] @ Wc^T + bc ; log_softmax   (tiny, VALU)
// ---------------------------------------------------------------------------

typedef __attribute__((ext_vector_type(16))) _Float16 v16h;
typedef __attribute__((ext_vector_type(8)))  _Float16 v8h;
typedef __attribute__((ext_vector_type(4)))  _Float16 v4h;
typedef __attribute__((ext_vector_type(8)))  float    v8f;

#define KB 256   // batch
#define KS 256   // seq len
#define KE 512   // embed dim
#define KH 1024  // hidden
#define KNG 4096 // 4*H gate width
#define LPITCH 34 // LDS row pitch (dwords); 8*34 % 64 == 16 -> conflict-free halves

__device__ __forceinline__ float sigmoidf(float x) { return 1.0f / (1.0f + __expf(-x)); }

// ---- prep: emb = tanh(embed[seq]) -> f16, layout [S][B][E] ---------------
__global__ __launch_bounds__(256) void embed_tanh_kernel(
    const int* __restrict__ seq, const float* __restrict__ embed,
    _Float16* __restrict__ X) {
  size_t id = ((size_t)blockIdx.x * 256 + threadIdx.x) * 4; // 4 elems/thread
  int t = (int)(id / ((size_t)KB * KE));
  int rem = (int)(id % ((size_t)KB * KE));
  int b = rem / KE, e = rem % KE;
  int tok = seq[(size_t)b * KS + t];
  float4 v = *(const float4*)(embed + (size_t)tok * KE + e);
  v4h o;
  o.x = (_Float16)tanhf(v.x); o.y = (_Float16)tanhf(v.y);
  o.z = (_Float16)tanhf(v.z); o.w = (_Float16)tanhf(v.w);
  *(v4h*)(X + id) = o;
}

// ---- prep: f32 -> f16 weight conversion ----------------------------------
__global__ __launch_bounds__(256) void cvt_f16_kernel(
    const float* __restrict__ src, _Float16* __restrict__ dst) {
  size_t id = ((size_t)blockIdx.x * 256 + threadIdx.x) * 4;
  float4 v = *(const float4*)(src + id);
  v4h o; o.x = (_Float16)v.x; o.y = (_Float16)v.y; o.z = (_Float16)v.z; o.w = (_Float16)v.w;
  *(v4h*)(dst + id) = o;
}

// ---- fused LSTM step: WMMA gate GEMM + LDS gate exchange + cell update ----
// Waves: wave = wm(2, M sub-block) + 2*gate(4).  Wave tile 32x32 at
// rows mBase+wm*32, cols gate*1024 + jBase.  Grid (1024/32, 256/64) = (32,4).
__global__ __launch_bounds__(256) void lstm_step_fused(
    const _Float16* __restrict__ X,     // [256, 512]  step input (f16)
    const _Float16* __restrict__ Wih,   // [4096, 512] row-major (f16)
    const _Float16* __restrict__ Hprev, // [256, 1024] f16
    const _Float16* __restrict__ Whh,   // [4096, 1024] row-major (f16)
    const float* __restrict__ bih, const float* __restrict__ bhh,
    float* __restrict__ c,              // [256, 1024] f32, in-place
    _Float16* __restrict__ Hnext,       // [256, 1024] f16 (ping-pong)
    float* __restrict__ hf32,           // [256, 1024] f32
    int useH, int zeroC) {
  __shared__ float lsG[4 * 64 * LPITCH];  // [gate][mm 0..63][jj 0..31], ~34 KB

  const int lane = threadIdx.x & 31;
  const int wave = threadIdx.x >> 5;
  const int wm   = wave & 1;
  const int gate = wave >> 1;
  const int mBase = blockIdx.y * 64;
  const int m0 = mBase + wm * 32;
  const int jBase = blockIdx.x * 32;
  const int n0 = gate * KH + jBase;
  const int l15 = lane & 15;
  const int hi  = lane >> 4;

  const v8f vzero = {0.f, 0.f, 0.f, 0.f, 0.f, 0.f, 0.f, 0.f};
  v8f acc[2][2];
  acc[0][0] = vzero; acc[0][1] = vzero; acc[1][0] = vzero; acc[1][1] = vzero;

  // A fragment (16x32 f16, M x K): lane l15 = row, VGPR0-3 hold K {hi*8..+7},
  // VGPR4-7 hold K {16+hi*8..+7}  (ISA 05_wmma layout).
  auto loadA = [&](const _Float16* A, int lda, int mB, int k0) -> v16h {
    const _Float16* p = A + (size_t)(mB + l15) * lda + k0 + hi * 8;
    v8h lo = *(const v8h*)p;
    v8h hh = *(const v8h*)(p + 16);
    return __builtin_shufflevector(lo, hh, 0,1,2,3,4,5,6,7,8,9,10,11,12,13,14,15);
  };
  // B fragment (32x16 f16, K x N): lane l15 = column n, elements 0-7 hold
  // K {hi*8..+7}, elements 8-15 hold K {16+hi*8..+7}.  B element (k,n) = W[n][k] (row-major W).
  auto loadB = [&](const _Float16* W, int ldw, int nB, int k0) -> v16h {
    const _Float16* p = W + (size_t)(nB + l15) * ldw + k0 + hi * 8;
    v8h lo = *(const v8h*)p;
    v8h hh = *(const v8h*)(p + 16);
    return __builtin_shufflevector(lo, hh, 0,1,2,3,4,5,6,7,8,9,10,11,12,13,14,15);
  };

  // Phase 1: input contribution, K = 512
  #pragma unroll 4
  for (int k0 = 0; k0 < KE; k0 += 32) {
    __builtin_prefetch(Wih + (size_t)(n0 + l15) * KE + k0 + 128, 0, 3);
    v16h a0 = loadA(X, KE, m0,      k0);
    v16h a1 = loadA(X, KE, m0 + 16, k0);
    v16h b0 = loadB(Wih, KE, n0,      k0);
    v16h b1 = loadB(Wih, KE, n0 + 16, k0);
    acc[0][0] = __builtin_amdgcn_wmma_f32_16x16x32_f16(false, a0, false, b0, (short)0, acc[0][0], false, false);
    acc[0][1] = __builtin_amdgcn_wmma_f32_16x16x32_f16(false, a0, false, b1, (short)0, acc[0][1], false, false);
    acc[1][0] = __builtin_amdgcn_wmma_f32_16x16x32_f16(false, a1, false, b0, (short)0, acc[1][0], false, false);
    acc[1][1] = __builtin_amdgcn_wmma_f32_16x16x32_f16(false, a1, false, b1, (short)0, acc[1][1], false, false);
  }
  // Phase 2: recurrent contribution, K = 1024
  if (useH) {
    #pragma unroll 4
    for (int k0 = 0; k0 < KH; k0 += 32) {
      __builtin_prefetch(Whh + (size_t)(n0 + l15) * KH + k0 + 128, 0, 3);
      v16h a0 = loadA(Hprev, KH, m0,      k0);
      v16h a1 = loadA(Hprev, KH, m0 + 16, k0);
      v16h b0 = loadB(Whh, KH, n0,      k0);
      v16h b1 = loadB(Whh, KH, n0 + 16, k0);
      acc[0][0] = __builtin_amdgcn_wmma_f32_16x16x32_f16(false, a0, false, b0, (short)0, acc[0][0], false, false);
      acc[0][1] = __builtin_amdgcn_wmma_f32_16x16x32_f16(false, a0, false, b1, (short)0, acc[0][1], false, false);
      acc[1][0] = __builtin_amdgcn_wmma_f32_16x16x32_f16(false, a1, false, b0, (short)0, acc[1][0], false, false);
      acc[1][1] = __builtin_amdgcn_wmma_f32_16x16x32_f16(false, a1, false, b1, (short)0, acc[1][1], false, false);
    }
  }

  // C/D 16x16 f32 layout: VGPR r, lanes<16 -> row +r, lanes>=16 -> row +r+8.
  #pragma unroll
  for (int i = 0; i < 2; ++i)
    #pragma unroll
    for (int j = 0; j < 2; ++j)
      #pragma unroll
      for (int r = 0; r < 8; ++r)
        lsG[(gate * 64 + wm * 32 + i * 16 + r + hi * 8) * LPITCH + j * 16 + l15] =
            acc[i][j][r];
  __syncthreads();

  // pointwise LSTM cell update: 64x32 elements, 8 per thread
  for (int e = threadIdx.x; e < 64 * 32; e += 256) {
    int mm = e >> 5, jj = e & 31;
    int m = mBase + mm;
    int j = jBase + jj;
    float gi = lsG[(0 * 64 + mm) * LPITCH + jj] + bih[j]        + bhh[j];
    float gf = lsG[(1 * 64 + mm) * LPITCH + jj] + bih[KH + j]   + bhh[KH + j];
    float gg = lsG[(2 * 64 + mm) * LPITCH + jj] + bih[2*KH + j] + bhh[2*KH + j];
    float go = lsG[(3 * 64 + mm) * LPITCH + jj] + bih[3*KH + j] + bhh[3*KH + j];
    int idx = m * KH + j;
    float cp = zeroC ? 0.0f : c[idx];
    float cn = sigmoidf(gf) * cp + sigmoidf(gi) * tanhf(gg);
    float hn = sigmoidf(go) * tanhf(cn);
    c[idx] = cn;
    Hnext[idx] = (_Float16)hn;
    hf32[idx] = hn;
  }
}

// ---- classifier: logits = [hf,hb]@Wc^T + bc ; log_softmax ----------------
__global__ __launch_bounds__(256) void classifier_kernel(
    const float* __restrict__ hf, const float* __restrict__ hb,
    const float* __restrict__ Wc, const float* __restrict__ bc,
    float* __restrict__ out) {
  int m = blockIdx.x, tid = threadIdx.x;
  float part[5] = {0.f, 0.f, 0.f, 0.f, 0.f};
  for (int k = tid; k < 2 * KH; k += 256) {
    float x = (k < KH) ? hf[(size_t)m * KH + k] : hb[(size_t)m * KH + k - KH];
    #pragma unroll
    for (int v = 0; v < 5; ++v) part[v] += x * Wc[(size_t)v * 2 * KH + k];
  }
  __shared__ float red[5][256];
  #pragma unroll
  for (int v = 0; v < 5; ++v) red[v][tid] = part[v];
  __syncthreads();
  for (int s = 128; s > 0; s >>= 1) {
    if (tid < s)
      #pragma unroll
      for (int v = 0; v < 5; ++v) red[v][tid] += red[v][tid + s];
    __syncthreads();
  }
  if (tid == 0) {
    float lg[5], mx = -1e30f;
    #pragma unroll
    for (int v = 0; v < 5; ++v) { lg[v] = red[v][0] + bc[v]; mx = fmaxf(mx, lg[v]); }
    float se = 0.f;
    #pragma unroll
    for (int v = 0; v < 5; ++v) se += __expf(lg[v] - mx);
    float lse = mx + __logf(se);
    #pragma unroll
    for (int v = 0; v < 5; ++v) out[(size_t)m * 5 + v] = lg[v] - lse;
  }
}

extern "C" void kernel_launch(void* const* d_in, const int* in_sizes, int n_in,
                              void* d_out, int out_size, void* d_ws, size_t ws_size,
                              hipStream_t stream) {
  const int*   seq   = (const int*)d_in[0];
  const float* embed = (const float*)d_in[1];
  const float* Wf_ih = (const float*)d_in[2];
  const float* Wf_hh = (const float*)d_in[3];
  const float* bf_ih = (const float*)d_in[4];
  const float* bf_hh = (const float*)d_in[5];
  const float* Wb_ih = (const float*)d_in[6];
  // d_in[7] Wb_hh, never needed: backward scan contributes only its 1st step.
  const float* bb_ih = (const float*)d_in[8];
  const float* bb_hh = (const float*)d_in[9];
  const float* Wc    = (const float*)d_in[10];
  const float* bc    = (const float*)d_in[11];
  float* out = (float*)d_out;

  char* ws = (char*)d_ws;
  size_t oX    = 0;                                // [S,B,E] f16   64 MB
  size_t oWih  = oX    + (size_t)KS * KB * KE * 2; // [4H,E] f16     4 MB
  size_t oWhh  = oWih  + (size_t)KNG * KE * 2;     // [4H,H] f16     8 MB
  size_t oWbih = oWhh  + (size_t)KNG * KH * 2;     // [4H,E] f16     4 MB
  size_t oH0   = oWbih + (size_t)KNG * KE * 2;     // [B,H]  f16 ping
  size_t oH1   = oH0   + (size_t)KB * KH * 2;      // [B,H]  f16 pong
  size_t oC    = oH1   + (size_t)KB * KH * 2;      // [B,H]  f32
  size_t oHf   = oC    + (size_t)KB * KH * 4;      // [B,H]  f32
  size_t oHb   = oHf   + (size_t)KB * KH * 4;      // [B,H]  f32

  _Float16* X      = (_Float16*)(ws + oX);
  _Float16* Wih16  = (_Float16*)(ws + oWih);
  _Float16* Whh16  = (_Float16*)(ws + oWhh);
  _Float16* Wbih16 = (_Float16*)(ws + oWbih);
  _Float16* Hping  = (_Float16*)(ws + oH0);
  _Float16* Hpong  = (_Float16*)(ws + oH1);
  float*    C      = (float*)(ws + oC);
  float*    Hf     = (float*)(ws + oHf);
  float*    Hb     = (float*)(ws + oHb);

  // zero LSTM state (deterministic across graph replays)
  hipMemsetAsync(Hping, 0, (size_t)KB * KH * 2, stream);
  hipMemsetAsync(C,     0, (size_t)KB * KH * 4, stream);

  // prep
  embed_tanh_kernel<<<(KS * KB * KE) / 4 / 256, 256, 0, stream>>>(seq, embed, X);
  cvt_f16_kernel<<<(KNG * KE) / 4 / 256, 256, 0, stream>>>(Wf_ih, Wih16);
  cvt_f16_kernel<<<(KNG * KH) / 4 / 256, 256, 0, stream>>>(Wf_hh, Whh16);
  cvt_f16_kernel<<<(KNG * KE) / 4 / 256, 256, 0, stream>>>(Wb_ih, Wbih16);

  dim3 stepGrid(KH / 32, KB / 64);  // (32, 4) = 128 WGs

  // forward LSTM: 256 sequential fused steps (h ping-pong: WGs read all of
  // h_prev while writing disjoint slices of h_next; c is per-WG in-place)
  _Float16* hin = Hping;
  _Float16* hout = Hpong;
  for (int t = 0; t < KS; ++t) {
    lstm_step_fused<<<stepGrid, 256, 0, stream>>>(
        X + (size_t)t * KB * KE, Wih16, hin, Whh16, bf_ih, bf_hh,
        C, hout, Hf, 1, 0);
    _Float16* tmp = hin; hin = hout; hout = tmp;
  }

  // backward LSTM collapses to a single step on emb[t=0], zero state
  lstm_step_fused<<<stepGrid, 256, 0, stream>>>(
      X, Wbih16, hin, Whh16, bb_ih, bb_hh, C, hout, Hb, 0, 1);

  // classifier + log_softmax
  classifier_kernel<<<KB, 256, 0, stream>>>(Hf, Hb, Wc, bc, out);
}
